// MoELayer_72524817760454
// MI455X (gfx1250) — matrix operand order repaired and checked by reading, not compiled, not run on hardware
//
#include <hip/hip_runtime.h>
#include <math.h>

// ---------------- problem constants ----------------
#define HDIM   1024
#define IDIM   4096
#define NEXP   8
#define NTOK   8192          // B*S
#define MTILE  32            // tokens per FFN block (2 WMMA row tiles)
#define BLK_PER_E 256        // NTOK / MTILE
#define ICHUNK 2048          // I processed per half (bounds LDS for h)
#define XS_DW  16384         // 32 tok * 1024 K * 2B  = 64KB  (A-layout x)
#define HS_DW  32768         // 32 tok * 2048 K * 2B  = 128KB (A-layout h half)
#define LDS_BYTES ((XS_DW + HS_DW) * 4)   // 192KB of the 320KB WGP LDS

// pre-swizzled bf16 weight tile grids (B-operand layout, 32B per lane slot)
#define UP_KC   32           // HDIM/32
#define UP_NT   256          // IDIM/16
#define DN_KC   128          // IDIM/32
#define DN_NT   64           // HDIM/16
#define UP_SLOTS (NEXP * UP_KC * UP_NT * 32)   // 2M lane-slots (32B each) = 64MB
#define DN_SLOTS (NEXP * DN_KC * DN_NT * 32)   // 2M lane-slots (32B each) = 64MB

// ws byte offsets
#define WS_COUNTS 0
#define WS_TLIST  1024
#define WS_WLIST  (WS_TLIST + NEXP * NTOK * 4)
#define WS_WUB    (1u << 20)                       // 1MB, 32B aligned
#define WS_WDB    (WS_WUB + (size_t)UP_SLOTS * 32)
#define WS_NEED   (WS_WDB + (size_t)DN_SLOTS * 32)

typedef __attribute__((ext_vector_type(16))) __bf16 v16bf;
typedef __attribute__((ext_vector_type(8)))  float  v8f;

// ---------------- helpers ----------------
// Single-instruction packed fp32->bf16 (RNE); builtin not exposed on gfx1250.
__device__ __forceinline__ unsigned cvt_pk_bf16(float lo, float hi) {
  unsigned r;
  asm("v_cvt_pk_bf16_f32 %0, %1, %2" : "=v"(r) : "v"(lo), "v"(hi));
  return r;
}

union V16U { v16bf v; unsigned u[8]; uint4 q[2]; };

// Raw fp32 B-tile fetch (32x16, K x N) from row-major weights, leading dim ld.
// B lane layout: lane = N | (khalf<<4); lanes 0-15 hold K0..15, lanes 16-31 K16..31.
__device__ __forceinline__ void load_b_raw(const float* W, int kbase, int col, int ld,
                                           float r[16]) {
  const int lane = threadIdx.x & 31;
  const float* p = W + (size_t)(kbase + ((lane >> 4) << 4)) * ld + col + (lane & 15);
  __builtin_prefetch(p + (size_t)64 * ld);   // global_prefetch_b8, next K tiles
#pragma unroll
  for (int d = 0; d < 16; ++d) r[d] = p[(size_t)d * ld];
}

__device__ __forceinline__ v16bf cvt_b16(const float r[16]) {
  V16U t;
#pragma unroll
  for (int d = 0; d < 8; ++d) t.u[d] = cvt_pk_bf16(r[2 * d], r[2 * d + 1]);
  return t.v;
}

// Pre-swizzled B tile: uniform base + 32-bit dword offset (saddr+voffset form),
// this lane's contiguous 32B -> 2x coalesced global_load_b128.
__device__ __forceinline__ v16bf load_b_pre(const unsigned* base, unsigned off_dw) {
  const uint4* p = (const uint4*)(base + off_dw + (unsigned)(threadIdx.x & 31) * 8u);
  V16U t;
  t.q[0] = p[0];
  t.q[1] = p[1];
  return t.v;
}

// A operand: this lane's 32B from LDS (already in WMMA A layout) -> 2x ds_load_b128
__device__ __forceinline__ v16bf load_a_lds(const unsigned* base) {
  V16U t;
  t.q[0] = ((const uint4*)base)[0];
  t.q[1] = ((const uint4*)base)[1];
  return t.v;
}

__device__ __forceinline__ v8f wmma_bf16(v16bf a, v16bf b, v8f c) {
  return __builtin_amdgcn_wmma_f32_16x16x32_bf16(false, a, false, b, (short)0, c, false, false);
}

// Branch-free tanh-form GELU: one v_exp_f32 + ~10 VALU, no EXEC divergence.
__device__ __forceinline__ float gelu_fast(float z) {
  float y = 0.7978845608028654f * z * (1.0f + 0.044715f * z * z);
  float t = __expf(-2.0f * fabsf(y));
  float th = (1.0f - t) / (1.0f + t);
  th = copysignf(th, y);
  return 0.5f * z * (1.0f + th);
}

// ---------------- kernels ----------------
__global__ void zero_fill(float* __restrict__ p, int n, int* __restrict__ counts) {
  int i = blockIdx.x * 256 + threadIdx.x;
  if (i < n) p[i] = 0.f;
  if (i < NEXP) counts[i] = 0;
}

// Convert + swizzle fp32 expert weights into bf16 WMMA-B tile layout (once per launch).
// Tile layout: [kc][nt][lane(32)][8 dw]; lane = n | (khalf<<4), dwords = packed K pairs.
__global__ void prep_weights(const float* __restrict__ up_w, const float* __restrict__ down_w,
                             unsigned* __restrict__ wub, unsigned* __restrict__ wdb) {
  unsigned s = blockIdx.x * 256 + threadIdx.x;
  if (s < UP_SLOTS) {
    const int lane = s & 31;
    const int nt = (s >> 5) & (UP_NT - 1);
    const int kc = (s >> 13) & (UP_KC - 1);
    const int e  = s >> 18;
    const float* p = up_w + (size_t)e * HDIM * IDIM +
                     (size_t)(kc * 32 + ((lane >> 4) << 4)) * IDIM + nt * 16 + (lane & 15);
    unsigned* dst = wub + (size_t)((((e * UP_KC) + kc) * UP_NT + nt) * 32 + lane) * 8;
    unsigned u[8];
#pragma unroll
    for (int d = 0; d < 8; ++d)
      u[d] = cvt_pk_bf16(p[(size_t)(2 * d) * IDIM], p[(size_t)(2 * d + 1) * IDIM]);
    ((uint4*)dst)[0] = make_uint4(u[0], u[1], u[2], u[3]);
    ((uint4*)dst)[1] = make_uint4(u[4], u[5], u[6], u[7]);
  } else {
    s -= UP_SLOTS;
    const int lane = s & 31;
    const int nt = (s >> 5) & (DN_NT - 1);
    const int kc = (s >> 11) & (DN_KC - 1);
    const int e  = s >> 18;
    const float* p = down_w + (size_t)e * IDIM * HDIM +
                     (size_t)(kc * 32 + ((lane >> 4) << 4)) * HDIM + nt * 16 + (lane & 15);
    unsigned* dst = wdb + (size_t)((((e * DN_KC) + kc) * DN_NT + nt) * 32 + lane) * 8;
    unsigned u[8];
#pragma unroll
    for (int d = 0; d < 8; ++d)
      u[d] = cvt_pk_bf16(p[(size_t)(2 * d) * HDIM], p[(size_t)(2 * d + 1) * HDIM]);
    ((uint4*)dst)[0] = make_uint4(u[0], u[1], u[2], u[3]);
    ((uint4*)dst)[1] = make_uint4(u[4], u[5], u[6], u[7]);
  }
}

__global__ void moe_router(const float* __restrict__ x, const float* __restrict__ gw,
                           float* __restrict__ usage, float* __restrict__ tope,
                           int* __restrict__ counts, int* __restrict__ tlist,
                           float* __restrict__ wlist) {
  const int lane = threadIdx.x & 31;
  const int t = blockIdx.x * 8 + (threadIdx.x >> 5);   // one wave32 per token
  float acc[NEXP];
#pragma unroll
  for (int e = 0; e < NEXP; ++e) acc[e] = 0.f;
  const float* xr = x + (size_t)t * HDIM;
  for (int h = lane; h < HDIM; h += 32) {
    const float xv = xr[h];
    const float* g = gw + h * NEXP;
#pragma unroll
    for (int e = 0; e < NEXP; ++e) acc[e] += xv * g[e];
  }
#pragma unroll
  for (int off = 16; off > 0; off >>= 1) {
#pragma unroll
    for (int e = 0; e < NEXP; ++e) acc[e] += __shfl_xor(acc[e], off, 32);
  }
  if (lane == 0) {
    float mx = acc[0];
#pragma unroll
    for (int e = 1; e < NEXP; ++e) mx = fmaxf(mx, acc[e]);
    float p[NEXP]; float s = 0.f;
#pragma unroll
    for (int e = 0; e < NEXP; ++e) { p[e] = __expf(acc[e] - mx); s += p[e]; }
    const float inv = 1.f / s;
#pragma unroll
    for (int e = 0; e < NEXP; ++e) {
      p[e] *= inv;
      atomicAdd(&usage[e], p[e] * (1.f / (float)NTOK));
    }
    int i0 = 0;
#pragma unroll
    for (int e = 1; e < NEXP; ++e) if (p[e] > p[i0]) i0 = e;
    int i1 = (i0 == 0) ? 1 : 0;
#pragma unroll
    for (int e = 0; e < NEXP; ++e) if (e != i0 && p[e] > p[i1]) i1 = e;
    const float wsum = p[i0] + p[i1];
    const float w0 = p[i0] / wsum, w1 = p[i1] / wsum;
    tope[t] = (float)i0;
    int pos0 = atomicAdd(&counts[i0], 1);
    tlist[i0 * NTOK + pos0] = t; wlist[i0 * NTOK + pos0] = w0;
    int pos1 = atomicAdd(&counts[i1], 1);
    tlist[i1 * NTOK + pos1] = t; wlist[i1 * NTOK + pos1] = w1;
  }
}

__global__ void loss_kernel(const float* __restrict__ usage, float* __restrict__ loss) {
  float s = 0.f;
#pragma unroll
  for (int e = 0; e < NEXP; ++e) s += usage[e] * usage[e];
  loss[0] = (float)NEXP * s;
}

// Fused grouped-GEMM expert FFN: out[t] += w * (gelu(x[t]@up + up_b) @ down + down_b)
// PRE=true : B from pre-swizzled bf16 tiles (2x global_load_b128 per tile)
// PRE=false: B from fp32 weights, converted in-flight (fallback when ws is small)
template <bool PRE>
__global__ void __launch_bounds__(256)
moe_ffn_t(const float* __restrict__ x,
          const float* __restrict__ up_w, const float* __restrict__ up_b,
          const float* __restrict__ down_w, const float* __restrict__ down_b,
          const unsigned* __restrict__ wub, const unsigned* __restrict__ wdb,
          const int* __restrict__ counts, const int* __restrict__ tlist,
          const float* __restrict__ wlist, float* __restrict__ out) {
  extern __shared__ unsigned smem[];
  unsigned* xs = smem;             // [mt][kc(32)][lane(32)][8 dw] : x, A-layout, K=1024
  unsigned* hs = smem + XS_DW;     // [mt][kc(64)][lane(32)][8 dw] : h half, A-layout, K=2048
  unsigned short* hs_us = (unsigned short*)hs;

  const int e   = blockIdx.x >> 8;          // BLK_PER_E == 256
  const int blk = blockIdx.x & (BLK_PER_E - 1);
  const int cnt = counts[e];
  const int start = blk * MTILE;
  if (start >= cnt) return;

  const int tid  = threadIdx.x;
  const int lane = tid & 31;
  const int wave = tid >> 5;                // 8 waves
  const int nn   = lane & 15;               // N / column index within tile
  const int lh   = lane >> 4;               // row-half selector for C/D layout

  __shared__ int   toks[MTILE];
  __shared__ float twt[MTILE];
  if (tid < MTILE) {
    int idx = start + tid;
    int cl  = idx < cnt ? idx : cnt - 1;    // pad with dup token, weight 0
    toks[tid] = tlist[e * NTOK + cl];
    twt[tid]  = idx < cnt ? wlist[e * NTOK + idx] : 0.f;
  }
  __syncthreads();

  // ---- stage gathered x rows into LDS in WMMA 16-bit A layout (bf16) ----
  for (int s = tid; s < 2 * 32 * 32; s += 256) {
    const int lslot = s & 31;
    const int kc    = (s >> 5) & 31;
    const int mt    = s >> 10;
    const int m     = lslot & 15;
    const int o0    = (lslot >> 4) ? 16 : 0;
    const float* row = x + (size_t)toks[mt * 16 + m] * HDIM + kc * 32;
    unsigned* dst = xs + (size_t)((mt * 32 + kc) * 32 + lslot) * 8;
    unsigned u[8];
#pragma unroll
    for (int d = 0; d < 4; ++d) u[d]     = cvt_pk_bf16(row[o0 + 2 * d],      row[o0 + 2 * d + 1]);
#pragma unroll
    for (int d = 0; d < 4; ++d) u[4 + d] = cvt_pk_bf16(row[8 + o0 + 2 * d], row[8 + o0 + 2 * d + 1]);
    ((uint4*)dst)[0] = make_uint4(u[0], u[1], u[2], u[3]);   // ds_store_b128
    ((uint4*)dst)[1] = make_uint4(u[4], u[5], u[6], u[7]);
  }
  __syncthreads();

  const float*    upw = up_w   + (size_t)e * HDIM * IDIM;
  const float*    dnw = down_w + (size_t)e * IDIM * HDIM;
  const unsigned* upt = wub + (size_t)e * UP_KC * UP_NT * 32 * 8;
  const unsigned* dnt = wdb + (size_t)e * DN_KC * DN_NT * 32 * 8;

  const v8f vzero = {0.f, 0.f, 0.f, 0.f, 0.f, 0.f, 0.f, 0.f};
  v8f acc[2][8];                            // [mtile][ntile] : 32 tok x 128 H cols per wave
#pragma unroll
  for (int mt = 0; mt < 2; ++mt)
#pragma unroll
    for (int nt = 0; nt < 8; ++nt) acc[mt][nt] = vzero;

  for (int half = 0; half < 2; ++half) {
    // ---- up projection + bias + GELU -> hs (stored in A layout for down proj) ----
    for (int nt = 0; nt < 16; ++nt) {
      const int i_local = wave * 256 + nt * 16;     // column base within this I-half
      const int i_glob  = half * ICHUNK + i_local;
      const unsigned ntg = (unsigned)(i_glob >> 4); // global up N-tile index
      v8f c0 = vzero, c1 = vzero;
      for (int kc = 0; kc < 32; kc += 2) {          // K = HDIM, 2-deep B pipeline
        v16bf bt0, bt1;
        if constexpr (PRE) {
          bt0 = load_b_pre(upt, ((unsigned)kc * UP_NT + ntg) * 256u);        // dwords/tile=256
          bt1 = load_b_pre(upt, ((unsigned)(kc + 1) * UP_NT + ntg) * 256u);
        } else {
          float r0[16], r1[16];
          load_b_raw(upw, kc * 32,       i_glob, IDIM, r0);
          load_b_raw(upw, (kc + 1) * 32, i_glob, IDIM, r1);
          bt0 = cvt_b16(r0);
          bt1 = cvt_b16(r1);
        }
        // hoist all four A tiles before the WMMAs (single LDS drain, no per-op stalls)
        v16bf a00 = load_a_lds(xs + (size_t)((0 * 32 + kc) * 32 + lane) * 8);
        v16bf a10 = load_a_lds(xs + (size_t)((1 * 32 + kc) * 32 + lane) * 8);
        v16bf a01 = load_a_lds(xs + (size_t)((0 * 32 + kc + 1) * 32 + lane) * 8);
        v16bf a11 = load_a_lds(xs + (size_t)((1 * 32 + kc + 1) * 32 + lane) * 8);
        c0 = wmma_bf16(a00, bt0, c0);
        c1 = wmma_bf16(a10, bt0, c1);
        c0 = wmma_bf16(a01, bt1, c0);
        c1 = wmma_bf16(a11, bt1, c1);
      }
      const float bias = up_b[e * IDIM + i_glob + nn];
      // scatter D (C/D layout) into hs A-layout position for K index k_local
      const int k_local = i_local + nn;
      const int kc2  = k_local >> 5;
      const int kk   = k_local & 31;
      const int bsel = (kk >> 4) & 1;
      const int dd   = (kk & 15) >> 1;
      const int elem = kk & 1;
#pragma unroll
      for (int v = 0; v < 8; ++v) {
        const int m  = v + lh * 8;
        const int tl = m | (bsel << 4);
        hs_us[((((0 * 64 + kc2) * 32 + tl) * 8 + dd) << 1) + elem] = (unsigned short)
            (cvt_pk_bf16(gelu_fast(c0[v] + bias), 0.f) & 0xffffu);
        hs_us[((((1 * 64 + kc2) * 32 + tl) * 8 + dd) << 1) + elem] = (unsigned short)
            (cvt_pk_bf16(gelu_fast(c1[v] + bias), 0.f) & 0xffffu);
      }
    }
    __syncthreads();
    // ---- down projection: accumulate over this I-half ----
#pragma unroll
    for (int nt = 0; nt < 8; ++nt) {
      const int h_col = wave * 128 + nt * 16;
      const unsigned ntg = (unsigned)(h_col >> 4);  // global down N-tile index
      for (int kc = 0; kc < 64; kc += 2) {          // K = ICHUNK, 2-deep B pipeline
        const int kcg = half * 64 + kc;             // global down K-chunk index
        v16bf bt0, bt1;
        if constexpr (PRE) {
          bt0 = load_b_pre(dnt, ((unsigned)kcg * DN_NT + ntg) * 256u);
          bt1 = load_b_pre(dnt, ((unsigned)(kcg + 1) * DN_NT + ntg) * 256u);
        } else {
          float r0[16], r1[16];
          load_b_raw(dnw, kcg * 32,       h_col, HDIM, r0);
          load_b_raw(dnw, (kcg + 1) * 32, h_col, HDIM, r1);
          bt0 = cvt_b16(r0);
          bt1 = cvt_b16(r1);
        }
        v16bf a00 = load_a_lds(hs + (size_t)((0 * 64 + kc) * 32 + lane) * 8);
        v16bf a10 = load_a_lds(hs + (size_t)((1 * 64 + kc) * 32 + lane) * 8);
        v16bf a01 = load_a_lds(hs + (size_t)((0 * 64 + kc + 1) * 32 + lane) * 8);
        v16bf a11 = load_a_lds(hs + (size_t)((1 * 64 + kc + 1) * 32 + lane) * 8);
        acc[0][nt] = wmma_bf16(a00, bt0, acc[0][nt]);
        acc[1][nt] = wmma_bf16(a10, bt0, acc[1][nt]);
        acc[0][nt] = wmma_bf16(a01, bt1, acc[0][nt]);
        acc[1][nt] = wmma_bf16(a11, bt1, acc[1][nt]);
      }
    }
    __syncthreads();                                // hs reused next half
  }

  // ---- epilogue: weighted combine scatter (token appears in <=2 expert lists) ----
#pragma unroll
  for (int mt = 0; mt < 2; ++mt)
#pragma unroll
    for (int nt = 0; nt < 8; ++nt) {
      const int h_col = wave * 128 + nt * 16 + nn;
      const float db = down_b[e * HDIM + h_col];
#pragma unroll
      for (int v = 0; v < 8; ++v) {
        const int m = mt * 16 + v + lh * 8;
        atomicAdd(&out[(size_t)toks[m] * HDIM + h_col], twt[m] * (acc[mt][nt][v] + db));
      }
    }
}

// ---------------- host launch ----------------
extern "C" void kernel_launch(void* const* d_in, const int* in_sizes, int n_in,
                              void* d_out, int out_size, void* d_ws, size_t ws_size,
                              hipStream_t stream) {
  const float* x      = (const float*)d_in[0];
  const float* gate_w = (const float*)d_in[1];
  const float* up_w   = (const float*)d_in[2];
  const float* up_b   = (const float*)d_in[3];
  const float* down_w = (const float*)d_in[4];
  const float* down_b = (const float*)d_in[5];

  float* out   = (float*)d_out;                       // [NTOK*HDIM]
  float* loss  = out + (size_t)NTOK * HDIM;           // [1]
  float* usage = loss + 1;                            // [NEXP]
  float* tope  = usage + NEXP;                        // [NTOK] (indices stored as float)

  char* ws = (char*)d_ws;
  int*      counts = (int*)(ws + WS_COUNTS);
  int*      tlist  = (int*)(ws + WS_TLIST);
  float*    wlist  = (float*)(ws + WS_WLIST);
  unsigned* wub    = (unsigned*)(ws + WS_WUB);
  unsigned* wdb    = (unsigned*)(ws + WS_WDB);

  const bool pre = (ws_size >= WS_NEED);              // constant per session -> deterministic

  const int nzero = NTOK * HDIM + 1 + NEXP;           // out + loss + usage (atomics target)
  zero_fill<<<(nzero + 255) / 256, 256, 0, stream>>>(out, nzero, counts);
  moe_router<<<NTOK / 8, 256, 0, stream>>>(x, gate_w, usage, tope, counts, tlist, wlist);
  loss_kernel<<<1, 1, 0, stream>>>(usage, loss);
  if (pre) {
    prep_weights<<<(UP_SLOTS + DN_SLOTS) / 256, 256, 0, stream>>>(up_w, down_w, wub, wdb);
    moe_ffn_t<true><<<NEXP * BLK_PER_E, 256, LDS_BYTES, stream>>>(
        x, up_w, up_b, down_w, down_b, wub, wdb, counts, tlist, wlist, out);
  } else {
    moe_ffn_t<false><<<NEXP * BLK_PER_E, 256, LDS_BYTES, stream>>>(
        x, up_w, up_b, down_w, down_b, wub, wdb, counts, tlist, wlist, out);
  }
}
